// PNet_28870770164326
// MI455X (gfx1250) — compile-verified, matrix operand order checked by reading, not run on hardware
//
#include <hip/hip_runtime.h>

typedef __attribute__((ext_vector_type(16))) _Float16     v16h;
typedef __attribute__((ext_vector_type(8)))  _Float16     v8h;
typedef __attribute__((ext_vector_type(8)))  float        v8f;
typedef __attribute__((ext_vector_type(4)))  unsigned int v4u;
typedef __attribute__((ext_vector_type(8)))  int          v8i;
typedef __attribute__((ext_vector_type(4)))  int          v4i;

#define PP    5
#define KW    11
#define CC    64
#define HH    256
#define WW    256
#define NP    64                 // points per batch
#define NB    2                  // batches
#define OW    246                // H - K + 1
#define KTOT  (KW * KW * CC)     // 7744, k = (dy*11+dx)*64 + c
#define MBLK  128                // x positions per workgroup
#define SLABR (MBLK - 1 + KW)    // 138 slab rows per dy
#define LROW  72                 // LDS row stride in halfs (64 + 8 pad -> no bank conflicts)

union AFrag { v16h v; v8h h[2]; };

// ------------------------------------------------------------------ TDM helper
// Builds a 2-group D# (2D tensor) and issues TENSOR_LOAD_TO_LDS (6-arg builtin).
// dims/strides in elements (data_size = 2 bytes). LDS rows get 4 dwords of pad
// per 32 dwords (one 64-half line) -> effective LDS row stride = 72 halfs.
__device__ __forceinline__ void tdm_load_2d(unsigned lds_addr, const void* gaddr,
                                            unsigned tile_d0, unsigned tile_d1,
                                            unsigned tensor_d0, unsigned tensor_d1,
                                            unsigned long long stride0) {
    const unsigned long long ga = (unsigned long long)gaddr;
    v4u g0;
    g0[0] = 1u;                                              // count=1, user descriptor
    g0[1] = lds_addr;                                        // lds_addr [63:32]
    g0[2] = (unsigned)(ga & 0xffffffffu);                    // global_addr low
    g0[3] = (unsigned)((ga >> 32) & 0x01ffffffu) | (2u << 30); // addr hi | type=2
    v8i g1;
    g1[0] = (int)((1u << 16) |                               // data_size = 2 bytes
                  (1u << 20) |                               // pad_enable
                  (4u << 22) |                               // pad_interval: 32 dwords
                  (3u << 25));                               // pad_amount: 4 dwords
    g1[1] = (int)((tensor_d0 & 0xffffu) << 16);              // tensor_dim0[15:0]
    g1[2] = (int)(((tensor_d0 >> 16) & 0xffffu) |
                  ((tensor_d1 & 0xffffu) << 16));            // dim0 hi | dim1 lo
    g1[3] = (int)(((tensor_d1 >> 16) & 0xffffu) |
                  ((tile_d0 & 0xffffu) << 16));              // dim1 hi | tile_dim0
    g1[4] = (int)(tile_d1 & 0xffffu);                        // tile_dim1 (tile_dim2=0)
    g1[5] = (int)(unsigned)(stride0 & 0xffffffffu);          // tensor_dim0_stride lo
    g1[6] = (int)(unsigned)((stride0 >> 32) & 0xffffu);      // stride hi (dim1_stride=0)
    g1[7] = 0;
    const v4i gz4 = {0, 0, 0, 0};                            // groups 2/3 unused (2D)
    const v8i gz8 = {0, 0, 0, 0, 0, 0, 0, 0};
    __builtin_amdgcn_tensor_load_to_lds(g0, g1, gz4, gz4, gz8, 0);
}

// ------------------------------------------------- NCHW f32 -> HWC f16 transform
// grid (W/16, H, NB), 256 threads
__global__ __launch_bounds__(256) void pnet_hwc(const float* __restrict__ in,
                                                _Float16* __restrict__ out) {
    __shared__ float tile[CC][17];
    const int x0 = blockIdx.x * 16, y = blockIdx.y, b = blockIdx.z;
    const float* src = in + (size_t)b * CC * HH * WW;
    _Float16* dst = out + (size_t)b * HH * WW * CC;
    for (int i = threadIdx.x; i < CC * 16; i += 256) {
        const int c = i >> 4, xl = i & 15;
        tile[c][xl] = src[((size_t)c * HH + y) * WW + x0 + xl];
    }
    __syncthreads();
    for (int i = threadIdx.x; i < CC * 16; i += 256) {
        const int xl = i >> 6, c = i & 63;
        dst[((size_t)(y * WW + x0 + xl)) * CC + c] = (_Float16)tile[c][xl];
    }
}

// ------------------------------------------------------------------ pack patches
// bpack[b][n][k] f16, k = (dy*11+dx)*64 + c  (reads HWC fm1 -> coalesced over c)
__global__ __launch_bounds__(256) void pnet_pack(const _Float16* __restrict__ fm1h,
                                                 const int* __restrict__ poi,
                                                 _Float16* __restrict__ bpack) {
    const int bn = blockIdx.x;               // 0..127
    const int b  = bn >> 6;
    const int py = poi[bn * 2 + 0];
    const int px = poi[bn * 2 + 1];
    const _Float16* src = fm1h + (size_t)b * HH * WW * CC;
    _Float16* dst = bpack + (size_t)bn * KTOT;
    for (int k = threadIdx.x; k < KTOT; k += 256) {
        const int c = k & 63, s = k >> 6;
        const int dy = s / KW, dx = s - dy * KW;
        dst[k] = src[((size_t)((py + dy) * WW + (px + dx))) * CC + c];
    }
}

// ---------------------------------------------------------------- zero output
__global__ __launch_bounds__(256) void pnet_zero(float* __restrict__ out, int total) {
    const int i = blockIdx.x * 256 + threadIdx.x;
    if (i < total) out[i] = 0.0f;
}

// ---------------------------------------------------------------- WMMA GEMM
// grid (2, 246, NB): x0 = 128*blockIdx.x, y = blockIdx.y, b = blockIdx.z.
// 256 threads = 8 wave32 waves; wave w owns M-tile w (16 x-positions) and all
// four N-tiles. Per dy: one TDM slab load of fm2-HWC; per dx: one TDM B chunk.
__global__ __launch_bounds__(256) void pnet_corr(const _Float16* __restrict__ fm2h,
                                                 const _Float16* __restrict__ bpack,
                                                 float* __restrict__ out) {
    __shared__ __align__(16) _Float16 Aslab[SLABR * LROW];   // 19,872 B
    __shared__ __align__(16) _Float16 Btile[NP * LROW];      //  9,216 B

    const int x0   = blockIdx.x * MBLK;
    const int y    = blockIdx.y;
    const int b    = blockIdx.z;
    const int tid  = threadIdx.x;
    const int lane = tid & 31;
    const int wave = tid >> 5;
    const int row  = lane & 15;
    const int khi  = lane >> 4;

    const _Float16* hwc = fm2h + (size_t)b * HH * WW * CC;
    const _Float16* bp  = bpack + (size_t)b * NP * KTOT;

    const unsigned ldsA = (unsigned)(unsigned long long)(void*)&Aslab[0];
    const unsigned ldsB = (unsigned)(unsigned long long)(void*)&Btile[0];

    v8f acc[4] = {};

    for (int dy = 0; dy < KW; ++dy) {
        const int flat0 = (y + dy) * WW + x0;         // flat spatial start of slab
        for (int dx = 0; dx < KW; ++dx) {
            __syncthreads();                          // LDS consumed by all waves
            if (wave == 0) {
                if (dx == 0)                          // fm2 slab: rows x0..x0+137 of (y+dy)
                    tdm_load_2d(ldsA, hwc + (size_t)flat0 * CC,
                                CC, SLABR, CC, (unsigned)(HH * WW - flat0), CC);
                // B chunk: 64 points x 64 channels at k0 (rows stride KTOT)
                const int k0 = (dy * KW + dx) * CC;
                tdm_load_2d(ldsB, bp + k0, CC, NP, CC, NP, KTOT);
                __builtin_amdgcn_s_wait_tensorcnt(0);
            }
            __syncthreads();                          // TDM data visible to all waves

            #pragma unroll
            for (int ks = 0; ks < 2; ++ks) {
                // A fragment: row M = wave*16+row, shifted dx rows into the slab
                const _Float16* ap =
                    &Aslab[(wave * 16 + row + dx) * LROW + ks * 32 + khi * 8];
                AFrag a;
                a.h[0] = *(const v8h*)ap;
                a.h[1] = *(const v8h*)(ap + 16);
                #pragma unroll
                for (int t = 0; t < 4; ++t) {
                    const _Float16* bq =
                        &Btile[(t * 16 + row) * LROW + ks * 32 + khi * 8];
                    AFrag bb;
                    bb.h[0] = *(const v8h*)bq;
                    bb.h[1] = *(const v8h*)(bq + 16);
                    acc[t] = __builtin_amdgcn_wmma_f32_16x16x32_f16(
                        false, a.v, false, bb.v, (short)0, acc[t], false, false);
                }
            }
        }
    }

    // store: lane holds col N = lane&15, VGPR r -> M = 8*khi + r (contiguous x)
    const int ncol  = lane & 15;
    const int mbase = wave * 16 + 8 * khi;
    #pragma unroll
    for (int t = 0; t < 4; ++t) {
        #pragma unroll
        for (int r = 0; r < 8; ++r) {
            const int x = x0 + mbase + r;
            if (x < OW) {
                const int n = t * 16 + ncol;
                out[(((size_t)b * NP + n) * HH + (y + PP)) * WW + (x + PP)] = acc[t][r];
            }
        }
    }
}

// ---------------------------------------------------------------- argmax
__global__ __launch_bounds__(256) void pnet_argmax(const float* __restrict__ out,
                                                   int* __restrict__ idx_out) {
    __shared__ float sval[256];
    __shared__ int   sidx[256];
    const int bn = blockIdx.x;
    const float* sm = out + (size_t)bn * HH * WW;
    float best = -3.402823e38f;
    int bidx = 0x7fffffff;
    for (int i = threadIdx.x; i < OW * OW; i += 256) {
        const int yy = i / OW, xx = i - yy * OW;
        const int pidx = (yy + PP) * WW + (xx + PP);  // padded row-major index
        const float v = sm[pidx];
        if (v > best || (v == best && pidx < bidx)) { best = v; bidx = pidx; }
    }
    sval[threadIdx.x] = best;
    sidx[threadIdx.x] = bidx;
    __syncthreads();
    for (int s = 128; s > 0; s >>= 1) {
        if (threadIdx.x < (unsigned)s) {
            const float v2 = sval[threadIdx.x + s];
            const int   i2 = sidx[threadIdx.x + s];
            if (v2 > sval[threadIdx.x] ||
                (v2 == sval[threadIdx.x] && i2 < sidx[threadIdx.x])) {
                sval[threadIdx.x] = v2;
                sidx[threadIdx.x] = i2;
            }
        }
        __syncthreads();
    }
    if (threadIdx.x == 0) {
        const float v = sval[0];
        int yy = 0, xx = 0;
        if (v > 0.0f) {                 // otherwise pad zero at flat index 0 wins
            const int pi = sidx[0];
            yy = pi / WW;
            xx = pi - yy * WW;
        }
        idx_out[bn * 2 + 0] = yy;
        idx_out[bn * 2 + 1] = xx;
    }
}

// ---------------------------------------------------------------- launch
extern "C" void kernel_launch(void* const* d_in, const int* in_sizes, int n_in,
                              void* d_out, int out_size, void* d_ws, size_t ws_size,
                              hipStream_t stream) {
    const float* fm1 = (const float*)d_in[0];
    const float* fm2 = (const float*)d_in[1];
    const int*   poi = (const int*)d_in[2];

    float* scores = (float*)d_out;                                 // [2,64,256,256]
    int*   maxidx = (int*)(scores + (size_t)NB * NP * HH * WW);    // [2,64,2] int32 bits

    // workspace layout (halfs): fm2h | fm1h | bpack  (~35.5 MB total)
    _Float16* fm2h  = (_Float16*)d_ws;
    _Float16* fm1h  = fm2h + (size_t)NB * HH * WW * CC;
    _Float16* bpack = fm1h + (size_t)NB * HH * WW * CC;

    dim3 tgrid(WW / 16, HH, NB);
    pnet_hwc<<<tgrid, 256, 0, stream>>>(fm2, fm2h);
    pnet_hwc<<<tgrid, 256, 0, stream>>>(fm1, fm1h);

    pnet_pack<<<NB * NP, 256, 0, stream>>>(fm1h, poi, bpack);

    const int total = NB * NP * HH * WW;
    pnet_zero<<<(total + 255) / 256, 256, 0, stream>>>(scores, total);

    dim3 grid(2, OW, NB);   // x0 in {0,128}, y rows, batches
    pnet_corr<<<grid, 256, 0, stream>>>(fm2h, bpack, scores);

    pnet_argmax<<<NB * NP, 256, 0, stream>>>(scores, maxidx);
}
